// AssociativeMemory_51376398795449
// MI455X (gfx1250) — matrix operand order repaired and checked by reading, not compiled, not run on hardware
//
#include <hip/hip_runtime.h>

// ---------------------------------------------------------------------------
// CDNA5 (gfx1250) fused associative-memory read.
//   out      = mean_h softmax( (query@Wq.T+bq)_h @ (mem_keys@Wk.T+bk)_h.T /8 )
//              @ (mem_values @ Wv.T) + bv
//   surprise = mean_v |target - out|
// Algebraic fold: (w @ V) @ Wv.T == w @ (V @ Wv.T)  -> precompute U once.
// All GEMMs use v_wmma_f32_16x16x32_bf16 (bf16 in, fp32 accumulate).
// Round-1 -> round-2 changes:
//   * Wq pre-converted to bf16 once (halves L2 bytes, removes hot-loop cvts)
//   * 32 query rows per workgroup: every B-tile feeds two A-tiles (2x reuse
//     of the L2-resident kbT / utb / Wqb operands)
// ---------------------------------------------------------------------------

typedef __attribute__((ext_vector_type(16))) __bf16 v16bf;
typedef __attribute__((ext_vector_type(8)))  __bf16 v8bf;
typedef __attribute__((ext_vector_type(8)))  float  v8f;

// ---- WMMA operand loaders -------------------------------------------------
// A-matrix 16x32 bf16 layout (ISA 7.12.2): lane l<16 -> row M=l, elems
// {K..K+7, K+16..K+23}; lane l>=16 -> row M=l-16, elems {K+8..K+15, K+24..K+31}.

__device__ __forceinline__ v16bf a_from_global_f32(const float* __restrict__ base,
                                                   int ld, int r0, int k0, int lid) {
  const int l = lid & 15, hi = (lid >> 4) & 1;
  const float* p0 = base + (size_t)(r0 + l) * ld + k0 + hi * 8;
  v16bf a;
#pragma unroll
  for (int e = 0; e < 8; ++e) { a[e] = (__bf16)p0[e]; a[8 + e] = (__bf16)p0[16 + e]; }
  return a;
}

__device__ __forceinline__ v16bf a_from_lds_bf16(const __bf16* base, int ld,
                                                 int k0, int lid) {
  const int l = lid & 15, hi = (lid >> 4) & 1;
  const __bf16* p0 = base + l * ld + k0 + hi * 8;
  v8bf c0 = *(const v8bf*)p0;
  v8bf c1 = *(const v8bf*)(p0 + 16);
  v16bf a;
#pragma unroll
  for (int e = 0; e < 8; ++e) { a[e] = c0[e]; a[8 + e] = c1[e]; }
  return a;
}

__device__ __forceinline__ v16bf a_from_lds_f32(const float* base, int ld,
                                                int k0, int lid) {
  const int l = lid & 15, hi = (lid >> 4) & 1;
  const float* p0 = base + l * ld + k0 + hi * 8;
  v16bf a;
#pragma unroll
  for (int e = 0; e < 8; ++e) { a[e] = (__bf16)p0[e]; a[8 + e] = (__bf16)p0[16 + e]; }
  return a;
}

// B-matrix 32x16 bf16 layout: lane n<16 -> col N=n, elems {K..K+7, K+16..K+23};
// lane n>=16 -> col N=n-16, elems {K+8..K+15, K+24..K+31}. Sources are N-major with
// K contiguous -> two 8-element contiguous runs per lane.

__device__ __forceinline__ v16bf b_from_global_f32(const float* __restrict__ base,
                                                   int ld, int n0, int k0, int lid) {
  const int n = lid & 15, half = (lid >> 4) & 1;
  const float* p = base + (size_t)(n0 + n) * ld + k0 + half * 8;
  v16bf b;
#pragma unroll
  for (int e = 0; e < 8; ++e) { b[e] = (__bf16)p[e]; b[8 + e] = (__bf16)p[16 + e]; }
  return b;
}

__device__ __forceinline__ v16bf b_from_global_bf16(const __bf16* __restrict__ base,
                                                    int ld, int n0, int k0, int lid) {
  const int n = lid & 15, half = (lid >> 4) & 1;
  const __bf16* p = base + (size_t)(n0 + n) * ld + k0 + half * 8;
  v8bf lo = *(const v8bf*)p;
  v8bf hi = *(const v8bf*)(p + 16);
  v16bf b;
#pragma unroll
  for (int e = 0; e < 8; ++e) { b[e] = lo[e]; b[8 + e] = hi[e]; }
  return b;
}

__device__ __forceinline__ v8f wmma_bf16(v16bf a, v16bf b, v8f c) {
  return __builtin_amdgcn_wmma_f32_16x16x32_bf16(false, a, false, b, (short)0, c,
                                                 false, false);
}

// ---------------------------------------------------------------------------
// Kernel 1: precompute (once per call)
//   blocks [0,128):   kbT[h][m][d] = bf16( (mem_keys @ Wk.T)[m, h*64+d] + bk )
//   blocks [128,256): utb[v][m]    = bf16( (Wv @ mem_values.T)[v, m] )
//   blocks [256,288): wqb[j][i]    = bf16( Wq[j][i] )   (already B-layout)
// ---------------------------------------------------------------------------
__global__ __launch_bounds__(256) void amem_prep_kernel(
    const float* __restrict__ mem_keys, const float* __restrict__ mem_values,
    const float* __restrict__ Wk, const float* __restrict__ bk,
    const float* __restrict__ Wv, const float* __restrict__ Wq,
    __bf16* __restrict__ kbT, __bf16* __restrict__ utb,
    __bf16* __restrict__ wqb) {
  const int t = threadIdx.x, lid = t & 31, wv = t >> 5;
  const int blk = blockIdx.x;

  if (blk >= 256) {  // task C: elementwise Wq -> bf16 (32 blocks x 256 thr x 32)
    const size_t base = ((size_t)(blk - 256) * 256 + t) * 32;
#pragma unroll
    for (int i = 0; i < 32; ++i) wqb[base + i] = (__bf16)Wq[base + i];
    return;
  }

  const bool taskA = blk < 128;
  const int id = (taskA ? blk : blk - 128) * 8 + wv;  // [0,1024)
  const int r0 = (id >> 5) << 4;  // A: m0 ; B: v0
  const int c0 = (id & 31) << 4;  // A: j0 ; B: m0

  const float* Abase = taskA ? mem_keys : Wv;
  const float* Bbase = taskA ? Wk : mem_values;

  v8f acc = {0.f, 0.f, 0.f, 0.f, 0.f, 0.f, 0.f, 0.f};
  for (int k0 = 0; k0 < 512; k0 += 32) {
    v16bf a = a_from_global_f32(Abase, 512, r0, k0, lid);
    v16bf b = b_from_global_f32(Bbase, 512, c0, k0, lid);
    acc = wmma_bf16(a, b, acc);
  }

  const int n = lid & 15, half = (lid >> 4) & 1;  // C/D: row = v + 8*half
  if (taskA) {
    const int j = c0 + n, h = j >> 6, d = j & 63;
    const float bias = bk[j];
#pragma unroll
    for (int v = 0; v < 8; ++v) {
      const int m = r0 + v + 8 * half;
      kbT[((size_t)h * 512 + m) * 64 + d] = (__bf16)(acc[v] + bias);
    }
  } else {
    const int m = c0 + n;
#pragma unroll
    for (int v = 0; v < 8; ++v) {
      const int vr = r0 + v + 8 * half;
      utb[(size_t)vr * 512 + m] = (__bf16)acc[v];
    }
  }
}

// ---------------------------------------------------------------------------
// Kernel 2: fused q-proj -> per-head scores+softmax -> head-mean -> out GEMM
// One workgroup (8 waves / 256 threads) owns 32 query rows (two 16-row
// A-tiles per wave; each B-tile is reused for both). Grid = 1024 WGs.
// LDS: qa 32KB (bf16 q), sbuf 64KB (scores / |diff|), wbuf 64KB (mean
// weights), 2x1KB reduction scratch  => 162KB of the 320KB WGP LDS.
// ---------------------------------------------------------------------------
__global__ __launch_bounds__(256) void amem_attn_kernel(
    const float* __restrict__ query, const float* __restrict__ target,
    const __bf16* __restrict__ wqb, const float* __restrict__ bq,
    const float* __restrict__ bv,
    const __bf16* __restrict__ kbT, const __bf16* __restrict__ utb,
    float* __restrict__ out, float* __restrict__ surprise) {
  __shared__ __bf16 qa[32 * 512];
  __shared__ float sbuf[32 * 512];
  __shared__ float wbuf[32 * 512];
  __shared__ float redA[32 * 8];
  __shared__ float redB[32 * 8];

  const int t = threadIdx.x, lid = t & 31, wv = t >> 5;
  const int b0 = blockIdx.x * 32;
  const int n = lid & 15, half = (lid >> 4) & 1;

  // ---- Phase 1: q = query @ Wq.T + bq  (wave wv -> out cols [wv*64, +64)) ---
  {
    v8f acc[2][4];
#pragma unroll
    for (int rt = 0; rt < 2; ++rt)
#pragma unroll
      for (int i = 0; i < 4; ++i)
        acc[rt][i] = (v8f){0.f, 0.f, 0.f, 0.f, 0.f, 0.f, 0.f, 0.f};
    for (int k0 = 0; k0 < 512; k0 += 32) {
      v16bf a0 = a_from_global_f32(query, 512, b0, k0, lid);
      v16bf a1 = a_from_global_f32(query, 512, b0 + 16, k0, lid);
#pragma unroll
      for (int nt = 0; nt < 4; ++nt) {
        v16bf b = b_from_global_bf16(wqb, 512, wv * 64 + nt * 16, k0, lid);
        acc[0][nt] = wmma_bf16(a0, b, acc[0][nt]);
        acc[1][nt] = wmma_bf16(a1, b, acc[1][nt]);
      }
    }
#pragma unroll
    for (int nt = 0; nt < 4; ++nt) {
      const int j = wv * 64 + nt * 16 + n;
      const float bias = bq[j];
#pragma unroll
      for (int rt = 0; rt < 2; ++rt)
#pragma unroll
        for (int v = 0; v < 8; ++v)
          qa[(rt * 16 + v + 8 * half) * 512 + j] = (__bf16)(acc[rt][nt][v] + bias);
    }
  }
#pragma unroll
  for (int i = 0; i < 64; ++i) wbuf[t * 64 + i] = 0.0f;  // zero mean-weights
  __syncthreads();

  // ---- Phase 2: per-head scores + exact softmax, accumulate mean into wbuf -
  const int r = t & 31, g = t >> 5;  // softmax mapping: row r, col group g*64
  for (int h = 0; h < 8; ++h) {
    {  // scores: wave wv covers memory slots m in [wv*64, +64), K = head dim 64
      const __bf16* kh = kbT + (size_t)h * 512 * 64;
      v8f acc[2][4];
#pragma unroll
      for (int rt = 0; rt < 2; ++rt)
#pragma unroll
        for (int i = 0; i < 4; ++i)
          acc[rt][i] = (v8f){0.f, 0.f, 0.f, 0.f, 0.f, 0.f, 0.f, 0.f};
#pragma unroll
      for (int kk = 0; kk < 64; kk += 32) {
        v16bf a0 = a_from_lds_bf16(qa, 512, h * 64 + kk, lid);
        v16bf a1 = a_from_lds_bf16(qa + 16 * 512, 512, h * 64 + kk, lid);
#pragma unroll
        for (int mt = 0; mt < 4; ++mt) {
          v16bf b = b_from_global_bf16(kh, 64, wv * 64 + mt * 16, kk, lid);
          acc[0][mt] = wmma_bf16(a0, b, acc[0][mt]);
          acc[1][mt] = wmma_bf16(a1, b, acc[1][mt]);
        }
      }
#pragma unroll
      for (int mt = 0; mt < 4; ++mt) {
        const int m = wv * 64 + mt * 16 + n;
#pragma unroll
        for (int rt = 0; rt < 2; ++rt)
#pragma unroll
          for (int v = 0; v < 8; ++v)
            sbuf[(rt * 16 + v + 8 * half) * 512 + m] =
                acc[rt][mt][v] * 0.125f;  // /sqrt(64)
      }
    }
    __syncthreads();
    // row max (row r, 8 groups of 64 columns)
    float pmax = -3.4e38f;
#pragma unroll
    for (int i = 0; i < 64; ++i) pmax = fmaxf(pmax, sbuf[r * 512 + g * 64 + i]);
    redA[r * 8 + g] = pmax;
    __syncthreads();
    float rmax = -3.4e38f;
#pragma unroll
    for (int i = 0; i < 8; ++i) rmax = fmaxf(rmax, redA[r * 8 + i]);
    // exp in place + row sum
    float psum = 0.0f;
#pragma unroll
    for (int i = 0; i < 64; ++i) {
      const int idx = r * 512 + g * 64 + i;
      const float e = __expf(sbuf[idx] - rmax);
      sbuf[idx] = e;
      psum += e;
    }
    redB[r * 8 + g] = psum;
    __syncthreads();
    float rsum = 0.0f;
#pragma unroll
    for (int i = 0; i < 8; ++i) rsum += redB[r * 8 + i];
    const float sc = 0.125f / rsum;  // normalize + 1/H head mean
#pragma unroll
    for (int i = 0; i < 64; ++i) {
      const int idx = r * 512 + g * 64 + i;
      wbuf[idx] += sbuf[idx] * sc;
    }
    __syncthreads();
  }

  // ---- Phase 3: out = wbuf @ U + bv ; |target-out| -> sbuf for reduction ----
  {
    v8f acc[2][4];
#pragma unroll
    for (int rt = 0; rt < 2; ++rt)
#pragma unroll
      for (int i = 0; i < 4; ++i)
        acc[rt][i] = (v8f){0.f, 0.f, 0.f, 0.f, 0.f, 0.f, 0.f, 0.f};
    for (int k0 = 0; k0 < 512; k0 += 32) {
      v16bf a0 = a_from_lds_f32(wbuf, 512, k0, lid);
      v16bf a1 = a_from_lds_f32(wbuf + 16 * 512, 512, k0, lid);
#pragma unroll
      for (int nt = 0; nt < 4; ++nt) {
        v16bf b = b_from_global_bf16(utb, 512, wv * 64 + nt * 16, k0, lid);
        acc[0][nt] = wmma_bf16(a0, b, acc[0][nt]);
        acc[1][nt] = wmma_bf16(a1, b, acc[1][nt]);
      }
    }
#pragma unroll
    for (int rt = 0; rt < 2; ++rt)
#pragma unroll
      for (int v = 0; v < 8; ++v) {
        const int row = rt * 16 + v + 8 * half;
        const int bb = b0 + row;
#pragma unroll
        for (int nt = 0; nt < 4; ++nt) {
          const int j = wv * 64 + nt * 16 + n;
          const float o = acc[rt][nt][v] + bv[j];
          out[(size_t)bb * 512 + j] = o;
          sbuf[row * 512 + j] = fabsf(target[(size_t)bb * 512 + j] - o);
        }
      }
  }
  __syncthreads();
  // deterministic serial reduction (no float atomics -> reproducible output)
  if (t < 32) {
    float ssum = 0.0f;
    for (int i = 0; i < 512; ++i) ssum += sbuf[t * 512 + i];
    surprise[b0 + t] = ssum * (1.0f / 512.0f);
  }
}

// ---------------------------------------------------------------------------
extern "C" void kernel_launch(void* const* d_in, const int* in_sizes, int n_in,
                              void* d_out, int out_size, void* d_ws, size_t ws_size,
                              hipStream_t stream) {
  const float* query      = (const float*)d_in[0];
  const float* target     = (const float*)d_in[1];
  const float* mem_keys   = (const float*)d_in[2];
  const float* mem_values = (const float*)d_in[3];
  const float* Wq = (const float*)d_in[4];
  const float* bq = (const float*)d_in[5];
  const float* Wk = (const float*)d_in[6];
  const float* bk = (const float*)d_in[7];
  const float* Wv = (const float*)d_in[8];
  const float* bv = (const float*)d_in[9];

  float* out      = (float*)d_out;
  float* surprise = out + (size_t)32768 * 512;

  __bf16* kbT = (__bf16*)d_ws;             // [8][512][64] bf16 = 512 KB
  __bf16* utb = kbT + (size_t)512 * 512;   // [512][512]   bf16 = 512 KB
  __bf16* wqb = utb + (size_t)512 * 512;   // [512][512]   bf16 = 512 KB

  amem_prep_kernel<<<288, 256, 0, stream>>>(mem_keys, mem_values, Wk, bk, Wv,
                                            Wq, kbT, utb, wqb);
  amem_attn_kernel<<<1024, 256, 0, stream>>>(query, target, wqb, bq, bv, kbT,
                                             utb, out, surprise);
}
